// GPRGNN_41386304864454
// MI455X (gfx1250) — matrix operand order repaired and checked by reading, not compiled, not run on hardware
//
#include <hip/hip_runtime.h>
#include <hip/hip_bf16.h>

// ---------------------------------------------------------------------------
// GPR-GNN on MI455X (gfx1250):
//   h1 = relu(x @ W1 + b1)   bf16x3 WMMA GEMM, B staged in LDS via
//   h2 = h1 @ W2 + b2        GLOBAL_LOAD_ASYNC_TO_LDS_B128 (double-buffered)
//   hops collapsed:          out[i] = g0*h2[i] + sum_k w_k*relu(h2[idx_k(i)])
// ---------------------------------------------------------------------------

typedef __attribute__((ext_vector_type(16))) __bf16 bf16x16;
typedef __attribute__((ext_vector_type(8)))  float  f32x8;

#define KHOPS 10

union BFrag {
    uint4   u[2];
    bf16x16 v;
};

__device__ __forceinline__ unsigned short f32_to_bf16_rne(float f) {
    unsigned u = __float_as_uint(f);
    unsigned r = u + 0x7FFFu + ((u >> 16) & 1u);
    return (unsigned short)(r >> 16);
}
__device__ __forceinline__ float bf16_to_f32(unsigned short h) {
    return __uint_as_float(((unsigned)h) << 16);
}
__device__ __forceinline__ unsigned pack2(unsigned short lo, unsigned short hi) {
    return (unsigned)lo | ((unsigned)hi << 16);
}
// LDS byte offset of a __shared__ object (addrspace(3) pointers are the raw
// LDS offset on amdgcn), needed as VDST for global_load_async_to_lds.
__device__ __forceinline__ unsigned lds_off(const void* p) {
    return (unsigned)(unsigned long long)(__attribute__((address_space(3))) const void*)p;
}

// ---------------------------------------------------------------------------
// Weight pre-pass: W [K x N] fp32 row-major  ->  Wt_hi/Wt_lo [N x K] bf16.
// Transposed so B fragments are contiguous 32B runs along K per column.
// ---------------------------------------------------------------------------
__global__ void wconvert_kernel(const float* __restrict__ W,
                                unsigned short* __restrict__ Th,
                                unsigned short* __restrict__ Tl,
                                int K, int N)
{
    int idx = blockIdx.x * blockDim.x + threadIdx.x;
    if (idx >= K * N) return;
    int k = idx / N;
    int n = idx - k * N;
    float w = W[idx];
    unsigned short h = f32_to_bf16_rne(w);
    unsigned short l = f32_to_bf16_rne(w - bf16_to_f32(h));
    Th[(size_t)n * K + k] = h;
    Tl[(size_t)n * K + k] = l;
}

// ---------------------------------------------------------------------------
// bf16x3 split-precision GEMM:  C = (relu?)(A[MxK] * W[KxN] + bias)
// 256 threads = 8 waves; wave w owns rows [m0, m0+16) x 128 cols.
// B tile (128 cols x 32 K, hi+lo) staged in LDS with async loads,
// double-buffered: issue next stage, s_wait_asynccnt 4, barrier, compute.
// ---------------------------------------------------------------------------
#define GEMM_BM   128
#define GEMM_WN   128
#define GEMM_NT   8
#define BPAD      40                    // shorts per column (32 + 8 pad)
#define MATBYTES  (128 * BPAD * 2)      // one hi or lo matrix     (10240 B)
#define BUFBYTES  (2 * MATBYTES)        // hi + lo per stage buffer (20480 B)

__global__ __launch_bounds__(256)
void gemm_bf16x3_kernel(const float* __restrict__ A,
                        const unsigned short* __restrict__ Bt_hi,
                        const unsigned short* __restrict__ Bt_lo,
                        const float* __restrict__ bias,
                        float* __restrict__ C,
                        int M, int K, int N, int do_relu)
{
    __shared__ unsigned short sB[2][2][128 * BPAD];   // [buf][hi/lo][col*BPAD+k]

    const int tid  = threadIdx.x;
    const int lane = tid & 31;
    const int wave = tid >> 5;
    const int hl   = lane >> 4;   // half-wave select
    const int l15  = lane & 15;

    const int m0 = blockIdx.y * GEMM_BM + wave * 16;
    const int n0 = blockIdx.x * GEMM_WN;

    // No early return: barriers below must be workgroup-uniform.
    int arow_i = m0 + l15;
    if (arow_i >= M) arow_i = 0;                 // clamp; stores are guarded
    const float* arow = A + (size_t)arow_i * K;

    // Fill-side addressing: thread -> (col fc, 16-short half fq); 4 async
    // b128 transfers per thread per stage (hi:2, lo:2).
    const int fc = tid >> 1;      // 0..127
    const int fq = tid & 1;       // 0..1
    const unsigned short* gh = Bt_hi + (size_t)(n0 + fc) * K + fq * 16;
    const unsigned short* gl = Bt_lo + (size_t)(n0 + fc) * K + fq * 16;
    const unsigned sbase = lds_off(&sB[0][0][0]);
    const unsigned dfill = (unsigned)(fc * BPAD + fq * 16) * 2u;

    auto stage = [&](int buf, int k0) {
        const unsigned short* srch = gh + k0;
        const unsigned short* srcl = gl + k0;
        unsigned dh = sbase + (unsigned)buf * BUFBYTES + dfill;
        unsigned dl = dh + MATBYTES;
        asm volatile("global_load_async_to_lds_b128 %0, %1, off"
                     :: "v"(dh), "v"(srch) : "memory");
        asm volatile("global_load_async_to_lds_b128 %0, %1, off offset:16"
                     :: "v"(dh), "v"(srch) : "memory");
        asm volatile("global_load_async_to_lds_b128 %0, %1, off"
                     :: "v"(dl), "v"(srcl) : "memory");
        asm volatile("global_load_async_to_lds_b128 %0, %1, off offset:16"
                     :: "v"(dl), "v"(srcl) : "memory");
    };

    f32x8 acc[GEMM_NT] = {};

    const int ksteps = K / 32;
    stage(0, 0);
    int buf = 0;

    for (int ks = 0; ks < ksteps; ++ks) {
        const int k0 = ks * 32;
        if (ks + 1 < ksteps) {
            stage(buf ^ 1, k0 + 32);
            // 4 newer transfers may stay in flight; the 4 for `buf` are done.
            asm volatile("s_wait_asynccnt 0x4" ::: "memory");
        } else {
            asm volatile("s_wait_asynccnt 0x0" ::: "memory");
        }
        __syncthreads();   // current buffer visible to all waves

        // ---- A fragment: lane (M=l15) needs K = k0+16*hl+e ----
        const float4* pa0 = (const float4*)(arow + k0 + hl * 16);
        const float4* pa1 = (const float4*)(arow + k0 + hl * 16 + 8);
        float4 q0 = pa0[0], q1 = pa0[1];
        float4 r0 = pa1[0], r1 = pa1[1];
        float fa[16] = { q0.x, q0.y, q0.z, q0.w, q1.x, q1.y, q1.z, q1.w,
                         r0.x, r0.y, r0.z, r0.w, r1.x, r1.y, r1.z, r1.w };
        unsigned short hs[16], ls[16];
        #pragma unroll
        for (int e = 0; e < 16; ++e) {
            hs[e] = f32_to_bf16_rne(fa[e]);
            ls[e] = f32_to_bf16_rne(fa[e] - bf16_to_f32(hs[e]));
        }
        BFrag ah, al;
        #pragma unroll
        for (int q = 0; q < 2; ++q) {
            ah.u[q] = make_uint4(pack2(hs[8*q+0], hs[8*q+1]), pack2(hs[8*q+2], hs[8*q+3]),
                                 pack2(hs[8*q+4], hs[8*q+5]), pack2(hs[8*q+6], hs[8*q+7]));
            al.u[q] = make_uint4(pack2(ls[8*q+0], ls[8*q+1]), pack2(ls[8*q+2], ls[8*q+3]),
                                 pack2(ls[8*q+4], ls[8*q+5]), pack2(ls[8*q+6], ls[8*q+7]));
        }

        // ---- 8 N-tiles from LDS (conflict-free: BPAD stride) ----
        #pragma unroll
        for (int t = 0; t < GEMM_NT; ++t) {
            const unsigned short* ph = &sB[buf][0][(t * 16 + l15) * BPAD + 16 * hl];
            const unsigned short* pl = &sB[buf][1][(t * 16 + l15) * BPAD + 16 * hl];
            BFrag bh, bl;
            bh.u[0] = ((const uint4*)ph)[0];
            bh.u[1] = ((const uint4*)ph)[1];
            bl.u[0] = ((const uint4*)pl)[0];
            bl.u[1] = ((const uint4*)pl)[1];
            // bf16x3: Ah*Bh + Al*Bh + Ah*Bl  (fp32 accumulate)
            acc[t] = __builtin_amdgcn_wmma_f32_16x16x32_bf16(
                         false, ah.v, false, bh.v, (short)0, acc[t], false, false);
            acc[t] = __builtin_amdgcn_wmma_f32_16x16x32_bf16(
                         false, al.v, false, bh.v, (short)0, acc[t], false, false);
            acc[t] = __builtin_amdgcn_wmma_f32_16x16x32_bf16(
                         false, ah.v, false, bl.v, (short)0, acc[t], false, false);
        }
        __syncthreads();   // reads done before next stage overwrites buf^1
        buf ^= 1;
    }

    // ---- epilogue: element r -> row m0 + 8*hl + r, col n0 + t*16 + l15 ----
    #pragma unroll
    for (int t = 0; t < GEMM_NT; ++t) {
        const int cn = n0 + t * 16 + l15;
        const float b = bias[cn];
        #pragma unroll
        for (int r = 0; r < 8; ++r) {
            const int cm = m0 + hl * 8 + r;
            float v = acc[t][r] + b;
            if (do_relu) v = fmaxf(v, 0.0f);
            if (cm < M) C[(size_t)cm * N + cn] = v;
        }
    }
}

// ---------------------------------------------------------------------------
// Graph-normalization pre-passes
// ---------------------------------------------------------------------------
__global__ void deg_kernel(const int* __restrict__ row, float* __restrict__ deg, int E)
{
    int e = blockIdx.x * blockDim.x + threadIdx.x;
    if (e < E) atomicAdd(&deg[row[e]], 1.0f);
}

__global__ void dis_kernel(float* __restrict__ deg, int N)
{
    int i = blockIdx.x * blockDim.x + threadIdx.x;
    if (i < N) {
        float d = deg[i];
        deg[i] = (d > 0.0f) ? rsqrtf(d) : 0.0f;
    }
}

__global__ void coef_kernel(const int* __restrict__ row, const int* __restrict__ col,
                            const float* __restrict__ dis, float* __restrict__ coef, int E)
{
    int e = blockIdx.x * blockDim.x + threadIdx.x;
    if (e < E) coef[e] = dis[row[e]] * dis[col[e]];
}

// ---------------------------------------------------------------------------
// Hop collapse. coef >= 0 and relu(c*relu(z)) == c*relu(z) for c>=0, so:
//   x_k[i] = (prod_{j<k} coef[col^j(i)]) * relu(h2[col^k(i)])
// ---------------------------------------------------------------------------
__global__ void chain_kernel(const int* __restrict__ col, const float* __restrict__ coef,
                             const float* __restrict__ gamma,
                             int* __restrict__ cidx, float* __restrict__ cw, int Nn)
{
    int i = blockIdx.x * blockDim.x + threadIdx.x;
    if (i >= Nn) return;
    int idx = i;
    float c = 1.0f;
    #pragma unroll
    for (int k = 1; k <= KHOPS; ++k) {
        c *= coef[idx];
        idx = col[idx];
        cidx[(size_t)i * KHOPS + (k - 1)] = idx;
        cw  [(size_t)i * KHOPS + (k - 1)] = gamma[k] * c;
    }
}

// out[i,:] = gamma[0]*h2[i,:] + sum_k w_k * relu(h2[idx_k,:])
__global__ void gather_accum_kernel(const float* __restrict__ h2,
                                    const int* __restrict__ cidx,
                                    const float* __restrict__ cw,
                                    const float* __restrict__ gamma,
                                    float* __restrict__ out, int Nn)
{
    const int C4 = 64;  // 256 floats / 4
    int tid = blockIdx.x * blockDim.x + threadIdx.x;
    if (tid >= Nn * C4) return;
    int i = tid >> 6;
    int c = tid & 63;
    const float4* base = (const float4*)h2;
    float4 v = base[(size_t)i * C4 + c];
    float g0 = gamma[0];
    float4 o;
    o.x = g0 * v.x; o.y = g0 * v.y; o.z = g0 * v.z; o.w = g0 * v.w;
    #pragma unroll
    for (int k = 0; k < KHOPS; ++k) {
        int   j = cidx[(size_t)i * KHOPS + k];
        float w = cw  [(size_t)i * KHOPS + k];
        float4 u = base[(size_t)j * C4 + c];
        o.x += w * fmaxf(u.x, 0.0f);
        o.y += w * fmaxf(u.y, 0.0f);
        o.z += w * fmaxf(u.z, 0.0f);
        o.w += w * fmaxf(u.w, 0.0f);
    }
    ((float4*)out)[(size_t)i * C4 + c] = o;
}

// ---------------------------------------------------------------------------
extern "C" void kernel_launch(void* const* d_in, const int* in_sizes, int n_in,
                              void* d_out, int out_size, void* d_ws, size_t ws_size,
                              hipStream_t stream)
{
    const float* x     = (const float*)d_in[0];
    const int*   eidx  = (const int*)d_in[1];
    const float* W1    = (const float*)d_in[2];
    const float* b1    = (const float*)d_in[3];
    const float* W2    = (const float*)d_in[4];
    const float* b2    = (const float*)d_in[5];
    const float* gamma = (const float*)d_in[6];

    const int IN_C = 512, HID_C = 512, OUT_C = 256;
    const int Nn = in_sizes[0] / IN_C;  // 200000
    const int E  = in_sizes[1] / 2;     // 200000
    const int* row = eidx;
    const int* col = eidx + E;

    // Workspace carve-out (~625 MB total)
    char* ws = (char*)d_ws;
    size_t off = 0;
    auto take = [&](size_t bytes) -> char* {
        char* p = ws + off;
        off = (off + bytes + 255) & ~(size_t)255;
        return p;
    };
    unsigned short* wt1h = (unsigned short*)take((size_t)IN_C  * HID_C * 2);
    unsigned short* wt1l = (unsigned short*)take((size_t)IN_C  * HID_C * 2);
    unsigned short* wt2h = (unsigned short*)take((size_t)HID_C * OUT_C * 2);
    unsigned short* wt2l = (unsigned short*)take((size_t)HID_C * OUT_C * 2);
    float* h1   = (float*)take((size_t)Nn * HID_C * 4);
    float* h2   = (float*)take((size_t)Nn * OUT_C * 4);
    float* dis  = (float*)take((size_t)Nn * 4);
    float* coef = (float*)take((size_t)E  * 4);
    int*   cidx = (int*)  take((size_t)Nn * KHOPS * 4);
    float* cw   = (float*)take((size_t)Nn * KHOPS * 4);

    hipMemsetAsync(dis, 0, (size_t)Nn * 4, stream);

    {   // weight split/transpose pre-passes (tiny)
        int tot1 = IN_C * HID_C;
        wconvert_kernel<<<(tot1 + 255) / 256, 256, 0, stream>>>(W1, wt1h, wt1l, IN_C, HID_C);
        int tot2 = HID_C * OUT_C;
        wconvert_kernel<<<(tot2 + 255) / 256, 256, 0, stream>>>(W2, wt2h, wt2l, HID_C, OUT_C);
    }

    deg_kernel <<<(E  + 255) / 256, 256, 0, stream>>>(row, dis, E);
    dis_kernel <<<(Nn + 255) / 256, 256, 0, stream>>>(dis, Nn);
    coef_kernel<<<(E  + 255) / 256, 256, 0, stream>>>(row, col, dis, coef, E);

    // GEMM1: x[200000x512] * W1 -> h1 (relu). N fastest => A L2 reuse.
    dim3 g1(HID_C / GEMM_WN, (Nn + GEMM_BM - 1) / GEMM_BM);
    gemm_bf16x3_kernel<<<g1, 256, 0, stream>>>(x, wt1h, wt1l, b1, h1, Nn, IN_C, HID_C, 1);

    // GEMM2: h1 * W2 -> h2 (no relu; raw h2 needed for gamma[0] term)
    dim3 g2(OUT_C / GEMM_WN, (Nn + GEMM_BM - 1) / GEMM_BM);
    gemm_bf16x3_kernel<<<g2, 256, 0, stream>>>(h1, wt2h, wt2l, b2, h2, Nn, HID_C, OUT_C, 0);

    chain_kernel<<<(Nn + 255) / 256, 256, 0, stream>>>(col, coef, gamma, cidx, cw, Nn);

    int tot = Nn * (OUT_C / 4);
    gather_accum_kernel<<<(tot + 255) / 256, 256, 0, stream>>>(h2, cidx, cw, gamma,
                                                               (float*)d_out, Nn);

    (void)n_in; (void)out_size; (void)ws_size;
}
